// StandardAttention_76312978915792
// MI455X (gfx1250) — matrix operand order repaired and checked by reading, not compiled, not run on hardware
//
#include <hip/hip_runtime.h>
#include <hip/hip_bf16.h>

typedef __attribute__((ext_vector_type(16))) __bf16 v16bf;
typedef __attribute__((ext_vector_type(8)))  __bf16 v8bf;
typedef __attribute__((ext_vector_type(8)))  float  v8f;
typedef int vint4 __attribute__((vector_size(16)));
typedef __attribute__((address_space(1))) vint4 g_vint4;   // global (device) int4
typedef __attribute__((address_space(3))) vint4 s_vint4;   // LDS int4

// Problem constants (B=1, N=4096, H=16, D=64)
#define NSEQ 4096
#define NH   16
#define DH   64
#define BM   64      // query rows per block (16 per wave x 4 waves)
#define BN   64      // keys per tile
#define LDK  72      // LDS row stride (elements) for K / P tiles (pad vs bank conflicts)
#define LDV  72      // LDS row stride (elements) for transposed V tile

#if __has_builtin(__builtin_amdgcn_global_load_async_to_lds_b128)
#define HAS_ASYNC_LDS 1
#else
#define HAS_ASYNC_LDS 0
#endif

__device__ __forceinline__ __bf16 f2bf(float f) { return (__bf16)f; }

// ---- DPP16 row_xmask butterfly reductions (stay within 16-lane row; wave32)
template <int CTRL>
__device__ __forceinline__ float dpp_xchg(float x) {
    int xi = __builtin_bit_cast(int, x);
    int yi = __builtin_amdgcn_update_dpp(0, xi, CTRL, 0xF, 0xF, true);
    return __builtin_bit_cast(float, yi);
}
__device__ __forceinline__ float redmax16(float x) {
    x = fmaxf(x, dpp_xchg<0x161>(x));   // ROW_XMASK:1
    x = fmaxf(x, dpp_xchg<0x162>(x));   // ROW_XMASK:2
    x = fmaxf(x, dpp_xchg<0x164>(x));   // ROW_XMASK:4
    x = fmaxf(x, dpp_xchg<0x168>(x));   // ROW_XMASK:8
    return x;
}
__device__ __forceinline__ float redsum16(float x) {
    x += dpp_xchg<0x161>(x);
    x += dpp_xchg<0x162>(x);
    x += dpp_xchg<0x164>(x);
    x += dpp_xchg<0x168>(x);
    return x;
}

// ======================================================================
// Kernel 1: one-shot repack  K -> bf16 Kb[h][n][d],  V -> bf16 Vt[h][d][n]
// ======================================================================
__global__ __launch_bounds__(256)
void repack_kv_kernel(const float* __restrict__ k, const float* __restrict__ v,
                      __bf16* __restrict__ kb, __bf16* __restrict__ vt) {
    int t = blockIdx.x * 256 + threadIdx.x;      // N*H*D/8 threads
    int d0 = (t & 7) * 8;                        // 8 consecutive d elements
    int nh = t >> 3;                             // n*H + h
    int h  = nh & (NH - 1);
    int n  = nh >> 4;

    const float* sk = k + (size_t)nh * DH + d0;
    __bf16 tk[8];
#pragma unroll
    for (int i = 0; i < 8; ++i) tk[i] = f2bf(sk[i]);
    *(v8bf*)(kb + ((size_t)h * NSEQ + n) * DH + d0) = *(const v8bf*)tk;

    const float* sv = v + (size_t)nh * DH + d0;
#pragma unroll
    for (int i = 0; i < 8; ++i)
        vt[((size_t)h * DH + d0 + i) * NSEQ + n] = f2bf(sv[i]);
}

// ======================================================================
// Kernel 2: flash attention, bf16 WMMA, async LDS staging
// ======================================================================
__global__ __launch_bounds__(128, 2)
void fa_bf16_wmma_kernel(const float* __restrict__ q,
                         const __bf16* __restrict__ kb,
                         const __bf16* __restrict__ vt,
                         float* __restrict__ out) {
    __shared__ __bf16 sK[BN][LDK];        // [key][d]
    __shared__ __bf16 sV[DH][LDV];        // [d][key]  (transposed V)
    __shared__ __bf16 sP[4][16][LDK];     // per-wave probability stripe [row][key]

    const int tid   = threadIdx.x;
    const int wave  = tid >> 5;
    const int lane  = tid & 31;
    const int lhalf = lane & 15;          // fragment column
    const int hsel  = lane >> 4;          // half-wave select
    const int kgb   = hsel * 8;           // A-fragment K group base

    const int numQ = NSEQ / BM;
    const int h    = blockIdx.x / numQ;
    const int q0   = (blockIdx.x % numQ) * BM;

    // ---- Load this wave's Q A-fragments (16 rows x 64 dims -> 2 chunks of 16x32)
    const int qrow = q0 + wave * 16 + lhalf;
    const float* qp = q + (size_t)qrow * (NH * DH) + h * DH;
    v16bf Qf[2];
#pragma unroll
    for (int c = 0; c < 2; ++c) {
#pragma unroll
        for (int i = 0; i < 8; ++i) {
            Qf[c][i]     = f2bf(qp[c * 32 + kgb + i]);        // K = kgb + i
            Qf[c][i + 8] = f2bf(qp[c * 32 + kgb + 16 + i]);   // K = kgb + 16 + i
        }
    }

    // ---- Online softmax state (C layout: VGPR r -> row r (+8 for upper half))
    float mrow[8], lrow[8];
    v8f O[4];
#pragma unroll
    for (int r = 0; r < 8; ++r) { mrow[r] = -__builtin_inff(); lrow[r] = 0.0f; }
#pragma unroll
    for (int dt = 0; dt < 4; ++dt)
#pragma unroll
        for (int e = 0; e < 8; ++e) O[dt][e] = 0.0f;

    const float cs = 0.125f * 1.44269504088896340736f; // (1/sqrt(64)) * log2(e)

    const __bf16* gk0 = kb + (size_t)h * NSEQ * DH;    // K tiles contiguous per head
    const __bf16* gv0 = vt + (size_t)h * DH * NSEQ;    // Vt rows stride NSEQ

    for (int kt = 0; kt < NSEQ; kt += BN) {
        __syncthreads();   // previous iteration done reading sK/sV

        // ---- Stage K tile (8KB contiguous) and Vt tile (64 rows x 128B) as pure copies
        const __bf16* gk = gk0 + (size_t)kt * DH;
        const __bf16* gv = gv0 + kt;
#if HAS_ASYNC_LDS
#pragma unroll
        for (int j = 0; j < 4; ++j) {
            int c   = tid + j * 128;       // 0..511 16-byte chunks
            int row = c >> 3;
            int col = c & 7;
            __builtin_amdgcn_global_load_async_to_lds_b128(
                (g_vint4*)(vint4*)(gk + c * 8),
                (s_vint4*)(&sK[row][col * 8]), 0, 0);
            __builtin_amdgcn_global_load_async_to_lds_b128(
                (g_vint4*)(vint4*)(gv + (size_t)row * NSEQ + col * 8),
                (s_vint4*)(&sV[row][col * 8]), 0, 0);
        }
#if __has_builtin(__builtin_amdgcn_s_wait_asynccnt)
        __builtin_amdgcn_s_wait_asynccnt(0);
#else
        asm volatile("s_wait_asynccnt 0x0" ::: "memory");
#endif
#else
#pragma unroll
        for (int j = 0; j < 4; ++j) {
            int c   = tid + j * 128;
            int row = c >> 3;
            int col = c & 7;
            *(v8bf*)(&sK[row][col * 8]) = *(const v8bf*)(gk + c * 8);
            *(v8bf*)(&sV[row][col * 8]) = *(const v8bf*)(gv + (size_t)row * NSEQ + col * 8);
        }
#endif
        __syncthreads();

        // ---- S = Q * K^T   (per wave: 16 x 64, fp32 accum; 8 WMMAs)
        v8f S[4];
#pragma unroll
        for (int nt = 0; nt < 4; ++nt)
#pragma unroll
            for (int e = 0; e < 8; ++e) S[nt][e] = 0.0f;

#pragma unroll
        for (int nt = 0; nt < 4; ++nt) {
#pragma unroll
            for (int c = 0; c < 2; ++c) {
                // B fragment: column n = key (nt*16 + lhalf); K-dim = d
                const __bf16* bp = &sK[nt * 16 + lhalf][hsel * 8 + c * 32];
                v8bf b0 = *(const v8bf*)bp;
                v8bf b1 = *(const v8bf*)(bp + 16);
                v16bf Bf;
#pragma unroll
                for (int i = 0; i < 8; ++i) { Bf[i] = b0[i]; Bf[i + 8] = b1[i]; }
                S[nt] = __builtin_amdgcn_wmma_f32_16x16x32_bf16(
                    false, Qf[c], false, Bf, (short)0, S[nt], false, false);
            }
        }

        // ---- Online softmax update + write P stripe to LDS (bf16)
#pragma unroll
        for (int r = 0; r < 8; ++r) {
            float mx = fmaxf(fmaxf(S[0][r], S[1][r]), fmaxf(S[2][r], S[3][r]));
            mx = redmax16(mx);
            float mnew  = fmaxf(mrow[r], cs * mx);
            float alpha = exp2f(mrow[r] - mnew);
            mrow[r] = mnew;

            float rs = 0.0f;
            int prow2 = r + hsel * 8;
#pragma unroll
            for (int nt = 0; nt < 4; ++nt) {
                float p = exp2f(S[nt][r] * cs - mnew);
                rs += p;
                sP[wave][prow2][nt * 16 + lhalf] = f2bf(p);
            }
            rs = redsum16(rs);
            lrow[r] = lrow[r] * alpha + rs;
#pragma unroll
            for (int dt = 0; dt < 4; ++dt) O[dt][r] *= alpha;
        }

        // ---- O += P * V   (A = P from per-wave LDS stripe, B = sV; 8 WMMAs)
#pragma unroll
        for (int c = 0; c < 2; ++c) {
            const __bf16* pp = &sP[wave][lhalf][c * 32 + kgb];
            v8bf a0 = *(const v8bf*)pp;
            v8bf a1 = *(const v8bf*)(pp + 16);
            v16bf Pf;
#pragma unroll
            for (int i = 0; i < 8; ++i) { Pf[i] = a0[i]; Pf[i + 8] = a1[i]; }
#pragma unroll
            for (int dt = 0; dt < 4; ++dt) {
                // B fragment: column n = output dim (dt*16 + lhalf); K-dim = key
                const __bf16* bp = &sV[dt * 16 + lhalf][hsel * 8 + c * 32];
                v8bf b0 = *(const v8bf*)bp;
                v8bf b1 = *(const v8bf*)(bp + 16);
                v16bf Bf;
#pragma unroll
                for (int i = 0; i < 8; ++i) { Bf[i] = b0[i]; Bf[i + 8] = b1[i]; }
                O[dt] = __builtin_amdgcn_wmma_f32_16x16x32_bf16(
                    false, Pf, false, Bf, (short)0, O[dt], false, false);
            }
        }
    }

    // ---- Normalize and store (out[b][n][h][d], fp32)
#pragma unroll
    for (int r = 0; r < 8; ++r) {
        float inv = 1.0f / lrow[r];
        int m = q0 + wave * 16 + r + hsel * 8;
        float* op = out + (size_t)m * (NH * DH) + h * DH + lhalf;
#pragma unroll
        for (int dt = 0; dt < 4; ++dt)
            op[dt * 16] = O[dt][r] * inv;
    }
}

extern "C" void kernel_launch(void* const* d_in, const int* in_sizes, int n_in,
                              void* d_out, int out_size, void* d_ws, size_t ws_size,
                              hipStream_t stream) {
    const float* q = (const float*)d_in[0];
    const float* k = (const float*)d_in[1];
    const float* v = (const float*)d_in[2];
    float* out = (float*)d_out;
    (void)in_sizes; (void)n_in; (void)out_size; (void)ws_size;

    // Workspace: Kb bf16 [H][N][D] then Vt bf16 [H][D][N]  (8 MB each)
    __bf16* kb = (__bf16*)d_ws;
    __bf16* vt = kb + (size_t)NH * NSEQ * DH;

    {
        dim3 grid((NSEQ * NH * DH / 8) / 256);   // 2048 blocks
        repack_kv_kernel<<<grid, 256, 0, stream>>>(k, v, kb, vt);
    }
    {
        dim3 grid(NH * (NSEQ / BM));             // 16 heads * 64 query tiles
        fa_bf16_wmma_kernel<<<grid, 128, 0, stream>>>(q, kb, vt, out);
    }
}
